// ForwardBackwardGNN_58067957842093
// MI455X (gfx1250) — matrix operand order repaired and checked by reading, not compiled, not run on hardware
//
#include <hip/hip_runtime.h>
#include <hip/hip_bf16.h>
#include <math.h>

#define NN 50000
#define EE 200000
#define LLEN 16
#define VV 30
#define DD 32
#define HH 64
#define HVD 214
#define HVP 224
#define RIDX_ 161
#define SID_ 53

// sched_barrier mask: allow ALU/VALU/SALU/DS-read/transcendental to cross;
// pin WMMA, DS-write, VMEM to their gate group.
#define SCHED_MASK 0x0507

typedef __attribute__((ext_vector_type(16))) _Float16 v16h;
typedef __attribute__((ext_vector_type(8)))  _Float16 v8h;
typedef __attribute__((ext_vector_type(8)))  float    v8f;

union U16h { v16h v; v8h h[2]; };

static __device__ __forceinline__ v8f wmma16(v16h a, v16h b, v8f c) {
  // D = A(16x32 f16) * B(32x16 f16) + C(16x16 f32)
  return __builtin_amdgcn_wmma_f32_16x16x32_f16(false, a, false, b, (short)0, c,
                                                false, false);
}

static __device__ __forceinline__ v8f vzero8() {
  v8f z;
#pragma unroll
  for (int r = 0; r < 8; ++r) z[r] = 0.f;
  return z;
}

// Branch-free fast sigmoid/tanh on the TRANS unit (v_exp_f32 + v_rcp_f32).
static __device__ __forceinline__ float sigf(float x) {
  return __builtin_amdgcn_rcpf(1.0f + __expf(-x));
}
static __device__ __forceinline__ float tanhfast(float x) {
  float e = __expf(2.0f * fabsf(x));            // +inf ok
  float t = 1.0f - 2.0f * __builtin_amdgcn_rcpf(e + 1.0f);
  return copysignf(t, x);
}

// A-operand tile (16x32 f16) from row-major [ld] f16 storage.
// lanes 0-15: M=lane, halves 0..7 = K kc..kc+7, halves 8..15 = K kc+16..kc+23
// lanes 16-31: M=lane-16, K offsets +8 (per ISA 7.12.2 16-bit A layout)
static __device__ __forceinline__ v16h load_a_rm(const _Float16* base, int ld,
                                                 int m0, int kc) {
  int l = threadIdx.x & 31;
  const _Float16* p = base + (size_t)(m0 + (l & 15)) * ld + kc + ((l >> 4) << 3);
  U16h u;
  u.h[0] = *(const v8h*)p;
  u.h[1] = *(const v8h*)(p + 16);
  return u.v;
}

// B-operand tile (32x16 f16) from W stored as [n][k] (so B[k][n] = W[n][k]).
// lanes 0-15 hold K=kc..kc+7,kc+16..kc+23 of column n0+lane; lanes 16-31 K offsets +8
static __device__ __forceinline__ v16h load_b_nk(const _Float16* base, int ldk,
                                                 int n0, int kc) {
  int l = threadIdx.x & 31;
  const _Float16* p = base + (size_t)(n0 + (l & 15)) * ldk + kc + ((l >> 4) << 3);
  U16h u;
  u.h[0] = *(const v8h*)p;
  u.h[1] = *(const v8h*)(p + 16);
  return u.v;
}

// ---------------------------------------------------------------------------
// Prep (shared across branches): P[v][g], hB table, f16 weight copies
// ---------------------------------------------------------------------------
__global__ void k_prep_shared(const float* __restrict__ embed,
                              const float* __restrict__ Wih_f,
                              const float* __restrict__ bih_f,
                              const float* __restrict__ bhh_f,
                              const float* __restrict__ Wih_r,
                              const float* __restrict__ bih_r,
                              const float* __restrict__ bhh_r,
                              const float* __restrict__ Whh_f,
                              const float* __restrict__ lin1w,
                              float* __restrict__ Pws,
                              float* __restrict__ hBws,
                              _Float16* __restrict__ Whh16,
                              _Float16* __restrict__ lin116) {
  int i = blockIdx.x * blockDim.x + threadIdx.x;
  const int TP = VV * 256;            // input-proj table
  const int TB = TP + VV * HH;        // reverse-hidden table
  const int TW = TB + 256 * HH;       // Whh f16
  const int TL = TW + 32 * 128;       // lin1 f16
  if (i < TP) {
    int v = i >> 8, g = i & 255;
    float s = bih_f[g] + bhh_f[g];
    for (int d = 0; d < DD; ++d) s += embed[v * DD + d] * Wih_f[g * DD + d];
    Pws[i] = s;
  } else if (i < TB) {
    int j = i - TP;
    int v = j >> 6, h = j & 63;
    float gi = bih_r[h] + bhh_r[h];
    float gg = bih_r[128 + h] + bhh_r[128 + h];
    float go = bih_r[192 + h] + bhh_r[192 + h];
    for (int d = 0; d < DD; ++d) {
      float ev = embed[v * DD + d];
      gi += ev * Wih_r[(h) * DD + d];
      gg += ev * Wih_r[(128 + h) * DD + d];
      go += ev * Wih_r[(192 + h) * DD + d];
    }
    float cn = sigf(gi) * tanhfast(gg);   // c0 = 0 -> forget path vanishes
    hBws[j] = sigf(go) * tanhfast(cn);
  } else if (i < TW) {
    int j = i - TB;
    Whh16[j] = (_Float16)Whh_f[j];     // [256][64] row-major == B as [n][k]
  } else if (i < TL) {
    int j = i - TW;
    lin116[j] = (_Float16)lin1w[j];    // [32][128] == B as [n][k]
  }
}

// ---------------------------------------------------------------------------
// Branch prep: pad GAT weights to f16 [224][224] as [n][k], pad 'a', init bufs
// ---------------------------------------------------------------------------
__global__ void k_prep_branch(const float* __restrict__ Wl,
                              const float* __restrict__ Wr,
                              const float* __restrict__ avec,
                              _Float16* __restrict__ Wl16,
                              _Float16* __restrict__ Wr16,
                              float* __restrict__ apad,
                              unsigned* __restrict__ emax,
                              float* __restrict__ denom,
                              float* __restrict__ conv) {
  const long T1 = (long)HVP * HVP;
  const long T2 = 2 * T1;
  const long T3 = T2 + HVP;
  const long T4 = T3 + NN;
  const long T5 = T4 + NN;
  const long total = T5 + (long)NN * HVD;
  for (long i = (long)blockIdx.x * blockDim.x + threadIdx.x; i < total;
       i += (long)gridDim.x * blockDim.x) {
    if (i < T1) {
      int n = (int)(i / HVP), k = (int)(i % HVP);
      Wl16[i] = (n < HVD && k < HVD) ? (_Float16)Wl[n * HVD + k] : (_Float16)0.f;
    } else if (i < T2) {
      long j = i - T1;
      int n = (int)(j / HVP), k = (int)(j % HVP);
      Wr16[j] = (n < HVD && k < HVD) ? (_Float16)Wr[n * HVD + k] : (_Float16)0.f;
    } else if (i < T3) {
      int k = (int)(i - T2);
      apad[k] = (k < HVD) ? avec[k] : 0.f;
    } else if (i < T4) {
      emax[i - T3] = 0x007FFFFFu;      // order-encoded -inf
    } else if (i < T5) {
      denom[i - T4] = 0.f;
    } else {
      conv[i - T5] = 0.f;
    }
  }
}

// ---------------------------------------------------------------------------
// LSTM recurrence: one wave owns 16 edges, h/c resident, 32 WMMA per step.
// Accumulators start at 0 (inline-0 C on first WMMA); the token-dependent
// input projection P is folded in at the activation stage. sched_barriers
// (masked) keep WMMA/DS-write/VMEM grouped per gate group while letting
// address math, activations and B-tile DS-reads float for pipelining.
// launch_bounds(128,2) caps VGPRs for >=2 waves/SIMD.
// ---------------------------------------------------------------------------
__global__ void __launch_bounds__(128, 2)
k_lstm(const int* __restrict__ regex, const float* __restrict__ Pws,
       const float* __restrict__ hBws, const _Float16* __restrict__ Whhws,
       _Float16* __restrict__ feat16) {
  extern __shared__ char smem[];
  float*    Psm   = (float*)smem;                              // 30*256 f32
  _Float16* Whhsm = (_Float16*)(smem + VV * 256 * 4);          // 256*64 f16
  float*    hBsm  = (float*)(smem + VV * 256 * 4 + 256 * HH * 2);
  char* wbase = smem + VV * 256 * 4 + 256 * HH * 2 + VV * HH * 4;
  int wave = threadIdx.x >> 5;
  int lane = threadIdx.x & 31;
  int*      toksm = (int*)(wbase + wave * 3072);               // 16*16 i32
  _Float16* hsm   = (_Float16*)(wbase + wave * 3072 + 1024);   // 16*64 f16

  for (int i = threadIdx.x; i < VV * 256; i += blockDim.x) Psm[i] = Pws[i];
  for (int i = threadIdx.x; i < 256 * HH; i += blockDim.x) Whhsm[i] = Whhws[i];
  for (int i = threadIdx.x; i < VV * HH; i += blockDim.x) hBsm[i] = hBws[i];
  __syncthreads();

  int tile = blockIdx.x * 4 + wave;     // 12500 tiles exactly
  int e0 = tile * 16;
  for (int i = lane; i < 256; i += 32) toksm[i] = regex[(size_t)e0 * LLEN + i];
  asm volatile("s_wait_dscnt 0x0" ::: "memory");

  int mo = (lane >> 4) << 3;            // C/D tile: M base for this half-wave
  int nl = lane & 15;                   // C/D tile: N = lane%16

  v16h hA0, hA1;
#pragma unroll
  for (int i = 0; i < 16; ++i) { hA0[i] = (_Float16)0.f; hA1[i] = (_Float16)0.f; }
  v8f cst[4];
#pragma unroll
  for (int j = 0; j < 4; ++j)
#pragma unroll
    for (int r = 0; r < 8; ++r) cst[j][r] = 0.f;

  for (int t = 0; t < LLEN; ++t) {
    // Tokens for this step depend only on (edge-row, t): hoist the 8 P-row
    // bases (pre-scaled to P rows) out of the gate-tile loops.
    int pbase[8];
#pragma unroll
    for (int r = 0; r < 8; ++r) pbase[r] = toksm[(mo + r) * 16 + t] * 256 + nl;

#pragma unroll
    for (int nh = 0; nh < 4; ++nh) {
      int colb = nh * 16;
      v8f g[4];
#pragma unroll
      for (int q = 0; q < 4; ++q) {
        v8f acc = vzero8();
        acc = wmma16(hA0, load_b_nk(Whhsm, HH, q * 64 + colb, 0), acc);
        acc = wmma16(hA1, load_b_nk(Whhsm, HH, q * 64 + colb, 32), acc);
        g[q] = acc;
      }
#pragma unroll
      for (int r = 0; r < 8; ++r) {
        float gi = g[0][r] + Psm[pbase[r] + colb];
        float gf = g[1][r] + Psm[pbase[r] + 64 + colb];
        float gg = g[2][r] + Psm[pbase[r] + 128 + colb];
        float go = g[3][r] + Psm[pbase[r] + 192 + colb];
        float cn = sigf(gf) * cst[nh][r] + sigf(gi) * tanhfast(gg);
        cst[nh][r] = cn;
        float hv = sigf(go) * tanhfast(cn);
        hsm[(mo + r) * HH + colb + nl] = (_Float16)hv;
      }
      // Keep WMMA scheduling window to one gate group (4 live accumulators),
      // but let ALU/addresses/DS-reads cross for pipelining.
      __builtin_amdgcn_sched_barrier(SCHED_MASK);
    }
    asm volatile("s_wait_dscnt 0x0" ::: "memory");
    hA0 = load_a_rm(hsm, HH, 0, 0);     // D-layout -> A-layout via LDS
    hA1 = load_a_rm(hsm, HH, 0, 32);
    __builtin_amdgcn_sched_barrier(SCHED_MASK);
  }
  asm volatile("s_wait_dscnt 0x0" ::: "memory");

  // feat = [hT | hB(last token)]  -> [16][128] f16
  for (int i = lane; i < 16 * HH; i += 32) {
    int m = i >> 6, j = i & 63;
    feat16[(size_t)(e0 + m) * 128 + j] = hsm[m * HH + j];
    feat16[(size_t)(e0 + m) * 128 + 64 + j] =
        (_Float16)hBsm[toksm[m * 16 + 15] * HH + j];
  }
}

// ---------------------------------------------------------------------------
// Edge MLP: lin1 (WMMA, K=128) + relu + lin2 dot + relu -> edge weight
// ---------------------------------------------------------------------------
__global__ void __launch_bounds__(128)
k_mlp(const _Float16* __restrict__ feat16, const _Float16* __restrict__ lin116,
      const float* __restrict__ lin1b, const float* __restrict__ lin2w,
      const float* __restrict__ lin2b, float* __restrict__ wE) {
  __shared__ _Float16 l1sm[32 * 128];
  __shared__ float l2sm[32];
  __shared__ float hs[4][16 * 32];
  for (int i = threadIdx.x; i < 32 * 128; i += blockDim.x) l1sm[i] = lin116[i];
  if (threadIdx.x < 32) l2sm[threadIdx.x] = lin2w[threadIdx.x];
  __syncthreads();

  int wave = threadIdx.x >> 5, lane = threadIdx.x & 31;
  int tile = blockIdx.x * 4 + wave;
  int e0 = tile * 16;
  int mo = (lane >> 4) << 3, nl = lane & 15;
  const _Float16* A = feat16 + (size_t)e0 * 128;

#pragma unroll
  for (int nt = 0; nt < 2; ++nt) {
    int n0 = nt * 16;
    v8f acc = vzero8();
#pragma unroll
    for (int kc = 0; kc < 128; kc += 32)
      acc = wmma16(load_a_rm(A, 128, 0, kc), load_b_nk(l1sm, 128, n0, kc), acc);
    float b = lin1b[n0 + nl];
#pragma unroll
    for (int r = 0; r < 8; ++r)
      hs[wave][(mo + r) * 32 + n0 + nl] = fmaxf(acc[r] + b, 0.f);
  }
  __syncthreads();
  if (lane < 16) {
    float s = lin2b[0];
#pragma unroll
    for (int j = 0; j < 32; ++j) s += hs[wave][lane * 32 + j] * l2sm[j];
    wE[e0 + lane] = fmaxf(s, 0.f);
  }
}

// ---------------------------------------------------------------------------
// Node feature copy + edge-weight scatter (set semantics, races arbitrary)
// ---------------------------------------------------------------------------
__global__ void k_copyx(const float* __restrict__ x, float* __restrict__ xmod) {
  const long total = (long)NN * HVD;
  for (long i = (long)blockIdx.x * blockDim.x + threadIdx.x; i < total;
       i += (long)gridDim.x * blockDim.x)
    xmod[i] = x[i];
}

__global__ void k_scatter(const float* __restrict__ x, const int* __restrict__ ei,
                          const float* __restrict__ wE, float* __restrict__ xmod,
                          int on_src) {
  int e = blockIdx.x * blockDim.x + threadIdx.x;
  if (e >= EE) return;
  int src = ei[e], dst = ei[EE + e];
  const float* row = x + (size_t)dst * HVD;
  float mx = row[0];
  int ti = 0;
  for (int j = 1; j < SID_; ++j) {
    float v = row[j];
    if (v > mx) { mx = v; ti = j; }
  }
  int r = on_src ? src : dst;
  xmod[(size_t)r * HVD + RIDX_ + ti] = wE[e];
}

__global__ void k_tof16(const float* __restrict__ xmod, _Float16* __restrict__ x16) {
  const long total = (long)NN * HVP;
  for (long i = (long)blockIdx.x * blockDim.x + threadIdx.x; i < total;
       i += (long)gridDim.x * blockDim.x) {
    int n = (int)(i / HVP), c = (int)(i % HVP);
    x16[i] = (c < HVD) ? (_Float16)xmod[(size_t)n * HVD + c] : (_Float16)0.f;
  }
}

// ---------------------------------------------------------------------------
// GAT projections: xl = x@Wl^T, xr = x@Wr^T  (WMMA GEMM, K padded to 224)
// ---------------------------------------------------------------------------
__global__ void __launch_bounds__(128)
k_gemm(const _Float16* __restrict__ x16, const _Float16* __restrict__ Wl16,
       const _Float16* __restrict__ Wr16, float* __restrict__ xl32,
       float* __restrict__ xr32) {
  int wave = threadIdx.x >> 5, lane = threadIdx.x & 31;
  int gid = blockIdx.x * 4 + wave;          // 2 * 3125 * 14 = 87500 tiles
  const int TPM = (NN / 16) * (HVP / 16);   // 43750
  int mat = gid / TPM;
  int rem = gid - mat * TPM;
  int tm = rem / (HVP / 16), tn = rem % (HVP / 16);
  const _Float16* B = mat ? Wr16 : Wl16;
  float* out = mat ? xr32 : xl32;
  int m0 = tm * 16, n0 = tn * 16;

  v8f acc = vzero8();
#pragma unroll
  for (int kc = 0; kc < HVP; kc += 32) {
    __builtin_prefetch(x16 + (size_t)(m0 + 16) * HVP + kc, 0, 1);
    acc = wmma16(load_a_rm(x16, HVP, m0, kc), load_b_nk(B, HVP, n0, kc), acc);
  }
  int mo = (lane >> 4) << 3, nl = lane & 15;
#pragma unroll
  for (int r = 0; r < 8; ++r)
    out[(size_t)(m0 + mo + r) * HVP + n0 + nl] = acc[r];
}

// ---------------------------------------------------------------------------
// Attention: e = lrelu(xl[src]+xr[dst])·a ; segment max/sum ; aggregate
// ---------------------------------------------------------------------------
__global__ void k_attnA(const int* __restrict__ ei, const float* __restrict__ xl,
                        const float* __restrict__ xr, const float* __restrict__ apad,
                        float* __restrict__ ews, unsigned* __restrict__ emax) {
  int wave = threadIdx.x >> 5, lane = threadIdx.x & 31;
  int e = blockIdx.x * 4 + wave;
  if (e >= EE) return;
  int src = ei[e], dst = ei[EE + e];
  float s = 0.f;
  for (int c = lane; c < HVP; c += 32) {
    float v = xl[(size_t)src * HVP + c] + xr[(size_t)dst * HVP + c];
    v = v > 0.f ? v : 0.2f * v;
    s += v * apad[c];
  }
  for (int off = 16; off > 0; off >>= 1) s += __shfl_xor(s, off, 32);
  if (lane == 0) {
    ews[e] = s;
    unsigned u = __float_as_uint(s);
    unsigned k = (u & 0x80000000u) ? ~u : (u | 0x80000000u);
    atomicMax(emax + dst, k);
  }
}

__global__ void k_attnB(const int* __restrict__ ei, float* __restrict__ ews,
                        const unsigned* __restrict__ emax,
                        float* __restrict__ denom) {
  int e = blockIdx.x * blockDim.x + threadIdx.x;
  if (e >= EE) return;
  int dst = ei[EE + e];
  unsigned k = emax[dst];
  unsigned u = (k & 0x80000000u) ? (k & 0x7FFFFFFFu) : ~k;
  float m = __uint_as_float(u);
  float expo = __expf(ews[e] - m);
  ews[e] = expo;
  atomicAdd(denom + dst, expo);
}

__global__ void k_attnC(const int* __restrict__ ei, const float* __restrict__ ews,
                        const float* __restrict__ denom,
                        const float* __restrict__ xl, float* __restrict__ conv) {
  int wave = threadIdx.x >> 5, lane = threadIdx.x & 31;
  int e = blockIdx.x * 4 + wave;
  if (e >= EE) return;
  int src = ei[e], dst = ei[EE + e];
  float alpha = ews[e] * __builtin_amdgcn_rcpf(fmaxf(denom[dst], 1e-16f));
  for (int c = lane; c < HVD; c += 32)
    atomicAdd(conv + (size_t)dst * HVD + c, alpha * xl[(size_t)src * HVP + c]);
}

__global__ void k_final(const float* __restrict__ conv, const float* __restrict__ bias,
                        const float* __restrict__ xmod, float* __restrict__ out,
                        int colbase) {
  const long total = (long)NN * HVD;
  for (long i = (long)blockIdx.x * blockDim.x + threadIdx.x; i < total;
       i += (long)gridDim.x * blockDim.x) {
    int n = (int)(i / HVD), c = (int)(i % HVD);
    out[(size_t)n * (2 * HVD) + colbase + c] = fmaxf(conv[i] + bias[c] + xmod[i], 0.f);
  }
}

// ---------------------------------------------------------------------------
extern "C" void kernel_launch(void* const* d_in, const int* in_sizes, int n_in,
                              void* d_out, int out_size, void* d_ws, size_t ws_size,
                              hipStream_t stream) {
  (void)in_sizes; (void)n_in; (void)out_size; (void)ws_size;
  const float* x_f   = (const float*)d_in[0];
  const float* x_b   = (const float*)d_in[1];
  const int*   ei_f  = (const int*)d_in[2];
  const int*   ei_b  = (const int*)d_in[3];
  const int*   rg_f  = (const int*)d_in[4];
  const int*   rg_b  = (const int*)d_in[5];
  const float* embed = (const float*)d_in[6];
  const float* Wih_f = (const float*)d_in[7];
  const float* Whh_f = (const float*)d_in[8];
  const float* bih_f = (const float*)d_in[9];
  const float* bhh_f = (const float*)d_in[10];
  const float* Wih_r = (const float*)d_in[11];
  const float* bih_r = (const float*)d_in[13];
  const float* bhh_r = (const float*)d_in[14];
  const float* lin1w = (const float*)d_in[15];
  const float* lin1b = (const float*)d_in[16];
  const float* lin2w = (const float*)d_in[17];
  const float* lin2b = (const float*)d_in[18];
  const float* gf_Wl = (const float*)d_in[19];
  const float* gf_Wr = (const float*)d_in[20];
  const float* gf_a  = (const float*)d_in[21];
  const float* gf_bi = (const float*)d_in[22];
  const float* gb_Wl = (const float*)d_in[23];
  const float* gb_Wr = (const float*)d_in[24];
  const float* gb_a  = (const float*)d_in[25];
  const float* gb_bi = (const float*)d_in[26];
  float* out = (float*)d_out;

  char* wsp = (char*)d_ws;
  size_t off = 0;
  auto alloc = [&](size_t bytes) -> char* {
    char* p = wsp + off;
    off += (bytes + 255) & ~(size_t)255;
    return p;
  };
  float*     Pws    = (float*)alloc(VV * 256 * 4);
  float*     hBws   = (float*)alloc(VV * HH * 4);
  _Float16*  Whh16  = (_Float16*)alloc(256 * HH * 2);
  _Float16*  lin116 = (_Float16*)alloc(32 * 128 * 2);
  _Float16*  Wl16   = (_Float16*)alloc((size_t)HVP * HVP * 2);
  _Float16*  Wr16   = (_Float16*)alloc((size_t)HVP * HVP * 2);
  float*     apad   = (float*)alloc(HVP * 4);
  unsigned*  emax   = (unsigned*)alloc((size_t)NN * 4);
  float*     denom  = (float*)alloc((size_t)NN * 4);
  float*     wEdge  = (float*)alloc((size_t)EE * 4);
  float*     ews    = (float*)alloc((size_t)EE * 4);
  float*     xmod   = (float*)alloc((size_t)NN * HVD * 4);
  float*     xl32   = (float*)alloc((size_t)NN * HVP * 4);
  float*     xr32   = (float*)alloc((size_t)NN * HVP * 4);
  float*     conv   = (float*)alloc((size_t)NN * HVD * 4);
  _Float16*  feat16 = (_Float16*)alloc((size_t)EE * 128 * 2);
  _Float16*  x16    = feat16;   // alias: feat16 dead before x16 is written

  const size_t lstm_lds = (size_t)VV * 256 * 4 + 256 * HH * 2 + VV * HH * 4 + 4 * 3072;

  // Shared prep
  {
    int total = VV * 256 + VV * HH + 256 * HH + 32 * 128;
    k_prep_shared<<<(total + 255) / 256, 256, 0, stream>>>(
        embed, Wih_f, bih_f, bhh_f, Wih_r, bih_r, bhh_r, Whh_f, lin1w,
        Pws, hBws, Whh16, lin116);
  }

  auto runBranch = [&](const float* x, const int* ei, const int* rg,
                       const float* Wl, const float* Wr, const float* av,
                       const float* bias, int on_src, int colbase) {
    k_prep_branch<<<4096, 256, 0, stream>>>(Wl, Wr, av, Wl16, Wr16, apad,
                                            emax, denom, conv);
    k_lstm<<<EE / 64, 128, lstm_lds, stream>>>(rg, Pws, hBws, Whh16, feat16);
    k_mlp<<<EE / 64, 128, 0, stream>>>(feat16, lin116, lin1b, lin2w, lin2b, wEdge);
    k_copyx<<<4096, 256, 0, stream>>>(x, xmod);
    k_scatter<<<(EE + 255) / 256, 256, 0, stream>>>(x, ei, wEdge, xmod, on_src);
    k_tof16<<<4096, 256, 0, stream>>>(xmod, x16);
    k_gemm<<<(2 * (NN / 16) * (HVP / 16)) / 4, 128, 0, stream>>>(x16, Wl16, Wr16,
                                                                 xl32, xr32);
    k_attnA<<<EE / 4, 128, 0, stream>>>(ei, xl32, xr32, apad, ews, emax);
    k_attnB<<<(EE + 255) / 256, 256, 0, stream>>>(ei, ews, emax, denom);
    k_attnC<<<EE / 4, 128, 0, stream>>>(ei, ews, denom, xl32, conv);
    k_final<<<4096, 256, 0, stream>>>(conv, bias, xmod, out, colbase);
  };

  runBranch(x_f, ei_f, rg_f, gf_Wl, gf_Wr, gf_a, gf_bi, /*on_src=*/1, /*col=*/0);
  runBranch(x_b, ei_b, rg_b, gb_Wl, gb_Wr, gb_a, gb_bi, /*on_src=*/0, /*col=*/HVD);
}
